// FpgnnModel_65798898974957
// MI455X (gfx1250) — matrix operand order repaired and checked by reading, not compiled, not run on hardware
//
#include <hip/hip_runtime.h>
#include <hip/hip_bf16.h>
#include <math.h>

#define N_CNT 50000
#define E_CNT 800000
#define HID 128

typedef __bf16 bf16_t;
typedef __attribute__((ext_vector_type(16))) __bf16 v16bf;
typedef __attribute__((ext_vector_type(8)))  __bf16 v8bf;
typedef __attribute__((ext_vector_type(8)))  float  v8f;

__device__ __forceinline__ float silu_f(float v) { return v / (1.f + __expf(-v)); }

// Assemble a WMMA A-operand fragment from two contiguous 16B LDS reads.
__device__ __forceinline__ v16bf load_afrag(const bf16_t* p) {
  v8bf lo = *(const v8bf*)p;
  v8bf hi = *(const v8bf*)(p + 16);
  return __builtin_shufflevector(lo, hi, 0,1,2,3,4,5,6,7,8,9,10,11,12,13,14,15);
}

// ---------------------------------------------------------------- prep kernels
__global__ void k_zero(float* __restrict__ p, long long n) {
  long long i  = (long long)blockIdx.x * blockDim.x + threadIdx.x;
  long long st = (long long)gridDim.x * blockDim.x;
  for (; i < n; i += st) p[i] = 0.f;
}

__global__ void k_cvt_bf16(const float* __restrict__ in, bf16_t* __restrict__ out, long long n) {
  long long i  = (long long)blockIdx.x * blockDim.x + threadIdx.x;
  long long st = (long long)gridDim.x * blockDim.x;
  for (; i < n; i += st) out[i] = (bf16_t)in[i];
}

// Pack a row-major (K,128) f32 weight matrix into WMMA-B blobs:
// blob(kc,nt) = 32 lanes x 16 halves; lane L holds col n = nt*16+(L&15),
// k = kc*32 + (L>=16 ? 8 : 0) + (i&7) + (i>=8 ? 16 : 0) for half i in [0,16).
__global__ void k_pack_b(const float* __restrict__ W, bf16_t* __restrict__ out, int K) {
  int t = blockIdx.x * blockDim.x + threadIdx.x;
  int total = K * 128;
  if (t >= total) return;
  int i    = t & 15;
  int L    = (t >> 4) & 31;
  int blob = t >> 9;
  int nt   = blob & 7;
  int kc   = blob >> 3;
  int k = kc * 32 + ((L >> 4) * 8) + (i & 7) + ((i >> 3) * 16);
  int n = nt * 16 + (L & 15);
  out[t] = (bf16_t)W[k * 128 + n];
}

// ---------------------------------------------------------------- edge kernel
// 2 waves per block, 32 edges per wave (two 16-row M tiles share each B fetch).
// m_ij (f32) overlays the gather buffer A (bf16): A is dead after GEMM1 +
// positional edge-MLP, which is hoisted before GEMM1.
struct EdgeSmem {
  union {
    alignas(16) bf16_t A[2][32][264];   // [x_src | x_dst] rows (bf16)
    alignas(16) float  M[2][32][132];   // m_ij (f32), written after A is dead
  } u;
  alignas(16) bf16_t H[2][32][136];     // hidden after layer1 (bf16)
  float D[2][32];                       // dists
  float G[2][32];                       // e_ij gate
};

__global__ __launch_bounds__(64) void k_edge(
    const bf16_t* __restrict__ xbf, const float* __restrict__ pos,
    const long long* __restrict__ eidx,
    const bf16_t* __restrict__ we1p, const float* __restrict__ We1,
    const float* __restrict__ be1,
    const bf16_t* __restrict__ we2p, const float* __restrict__ be2,
    const float* __restrict__ Winf, const float* __restrict__ binf,
    const float* __restrict__ Wp1, const float* __restrict__ bp1,
    const float* __restrict__ Wp2, const float* __restrict__ bp2,
    float* __restrict__ m_i, float* __restrict__ m_i_pos)
{
  __shared__ EdgeSmem s;
  const int lane = threadIdx.x & 31;
  const int wid  = threadIdx.x >> 5;
  const long long e0 = ((long long)blockIdx.x * 2 + wid) * 32;

  __builtin_prefetch(we1p + lane * 16, 0, 3);   // global_prefetch_b8: warm weights
  __builtin_prefetch(we2p + lane * 16, 0, 3);

  // ---- stage gathered bf16 features: row j = [x[src] (128) | x[dst] (128)] ----
  for (int j = 0; j < 32; ++j) {
    long long e = e0 + j;
    long long es = 0, ee = 0;
    if (e < E_CNT) { es = eidx[2 * e]; ee = eidx[2 * e + 1]; }
    const unsigned* ps = (const unsigned*)(xbf + es * HID);
    const unsigned* pd = (const unsigned*)(xbf + ee * HID);
    unsigned* dst = (unsigned*)(&s.u.A[wid][j][0]);
    dst[lane * 2 + 0]      = ps[lane * 2 + 0];
    dst[lane * 2 + 1]      = ps[lane * 2 + 1];
    dst[64 + lane * 2 + 0] = pd[lane * 2 + 0];
    dst[64 + lane * 2 + 1] = pd[lane * 2 + 1];
  }
  {
    long long e = e0 + lane;
    float dist = 0.f;
    if (e < E_CNT) {
      long long es = eidx[2 * e], ee = eidx[2 * e + 1];
      float dx = pos[es * 3 + 0] - pos[ee * 3 + 0];
      float dy = pos[es * 3 + 1] - pos[ee * 3 + 1];
      float dz = pos[es * 3 + 2] - pos[ee * 3 + 2];
      dist = sqrtf(dx * dx + dy * dy + dz * dz);
    }
    s.D[wid][lane] = dist;
  }
  __syncthreads();

  // ---- positional edge MLP (VALU, one lane per edge) — consumes A, so do it
  //      before m_ij overlays A ----
  {
    long long e = e0 + lane;
    if (e < E_CNT) {
      float t0 = bp1[0], t1 = bp1[1], t2 = bp1[2];
      for (int c2 = 0; c2 < 256; ++c2) {
        float f = (float)s.u.A[wid][lane][c2];
        t0 += f * Wp1[c2 * 3 + 0];
        t1 += f * Wp1[c2 * 3 + 1];
        t2 += f * Wp1[c2 * 3 + 2];
      }
      float dist = s.D[wid][lane];
      t0 += dist * Wp1[256 * 3 + 0];
      t1 += dist * Wp1[256 * 3 + 1];
      t2 += dist * Wp1[256 * 3 + 2];
      t0 = silu_f(t0); t1 = silu_f(t1); t2 = silu_f(t2);
      float u0 = silu_f(bp2[0] + t0 * Wp2[0] + t1 * Wp2[3] + t2 * Wp2[6]);
      float u1 = silu_f(bp2[1] + t0 * Wp2[1] + t1 * Wp2[4] + t2 * Wp2[7]);
      float u2 = silu_f(bp2[2] + t0 * Wp2[2] + t1 * Wp2[5] + t2 * Wp2[8]);
      long long src = eidx[2 * e];
      float sc = dist / (float)N_CNT;
      atomicAdd(&m_i_pos[src * 3 + 0], sc * u0);
      atomicAdd(&m_i_pos[src * 3 + 1], sc * u1);
      atomicAdd(&m_i_pos[src * 3 + 2], sc * u2);
    }
  }

  const int rlo  = lane & 15;
  const int koff = (lane >> 4) * 8;     // A-operand half-lane K offset
  const int n0   = lane & 15;           // C layout: N = lane&15
  const int mhi  = (lane >> 4) * 8;     // C layout: M base = 8*(lane>=16)
  const bf16_t* b1base = we1p + lane * 16;
  const bf16_t* b2base = we2p + lane * 16;

  // ---- GEMM1: (32x256) @ (256x128); software-pipelined A (LDS) and B (L2) ----
  v8f acc0[8], acc1[8];
  #pragma unroll
  for (int nt = 0; nt < 8; ++nt)
    #pragma unroll
    for (int r = 0; r < 8; ++r) { acc0[nt][r] = 0.f; acc1[nt][r] = 0.f; }

  {
    v16bf a0 = load_afrag(&s.u.A[wid][rlo][koff]);
    v16bf a1 = load_afrag(&s.u.A[wid][16 + rlo][koff]);
    v16bf bc = *(const v16bf*)(b1base);
    #pragma unroll
    for (int kc = 0; kc < 8; ++kc) {
      v16bf a0n = a0, a1n = a1;
      if (kc + 1 < 8) {
        a0n = load_afrag(&s.u.A[wid][rlo][(kc + 1) * 32 + koff]);
        a1n = load_afrag(&s.u.A[wid][16 + rlo][(kc + 1) * 32 + koff]);
      }
      #pragma unroll
      for (int nt = 0; nt < 8; ++nt) {
        const int idx = kc * 8 + nt;
        v16bf bn = bc;
        if (idx + 1 < 64) bn = *(const v16bf*)(b1base + (idx + 1) * 512);
        acc0[nt] = __builtin_amdgcn_wmma_f32_16x16x32_bf16(false, a0, false, bc,
                                                           (short)0, acc0[nt], false, false);
        acc1[nt] = __builtin_amdgcn_wmma_f32_16x16x32_bf16(false, a1, false, bc,
                                                           (short)0, acc1[nt], false, false);
        bc = bn;
      }
      a0 = a0n; a1 = a1n;
    }
  }
  __syncthreads();   // everyone done reading A (pos-MLP + GEMM1)

  // rank-1 dist column (row 256 of We1) + bias + SiLU -> bf16 LDS
  #pragma unroll
  for (int nt = 0; nt < 8; ++nt) {
    const int n = nt * 16 + n0;
    const float w256 = We1[256 * HID + n];
    const float bias = be1[n];
    #pragma unroll
    for (int r = 0; r < 8; ++r) {
      const int m0 = mhi + r;
      const int m1 = 16 + mhi + r;
      s.H[wid][m0][n] = (bf16_t)silu_f(acc0[nt][r] + s.D[wid][m0] * w256 + bias);
      s.H[wid][m1][n] = (bf16_t)silu_f(acc1[nt][r] + s.D[wid][m1] * w256 + bias);
    }
  }
  __syncthreads();

  // ---- GEMM2: (32x128) @ (128x128), same pipelining ----
  v8f acc2a[8], acc2b[8];
  #pragma unroll
  for (int nt = 0; nt < 8; ++nt)
    #pragma unroll
    for (int r = 0; r < 8; ++r) { acc2a[nt][r] = 0.f; acc2b[nt][r] = 0.f; }

  {
    v16bf a0 = load_afrag(&s.H[wid][rlo][koff]);
    v16bf a1 = load_afrag(&s.H[wid][16 + rlo][koff]);
    v16bf bc = *(const v16bf*)(b2base);
    #pragma unroll
    for (int kc = 0; kc < 4; ++kc) {
      v16bf a0n = a0, a1n = a1;
      if (kc + 1 < 4) {
        a0n = load_afrag(&s.H[wid][rlo][(kc + 1) * 32 + koff]);
        a1n = load_afrag(&s.H[wid][16 + rlo][(kc + 1) * 32 + koff]);
      }
      #pragma unroll
      for (int nt = 0; nt < 8; ++nt) {
        const int idx = kc * 8 + nt;
        v16bf bn = bc;
        if (idx + 1 < 32) bn = *(const v16bf*)(b2base + (idx + 1) * 512);
        acc2a[nt] = __builtin_amdgcn_wmma_f32_16x16x32_bf16(false, a0, false, bc,
                                                            (short)0, acc2a[nt], false, false);
        acc2b[nt] = __builtin_amdgcn_wmma_f32_16x16x32_bf16(false, a1, false, bc,
                                                            (short)0, acc2b[nt], false, false);
        bc = bn;
      }
      a0 = a0n; a1 = a1n;
    }
  }
  // m_ij = silu(. + be2) -> f32 LDS (overlays A, which is dead now)
  #pragma unroll
  for (int nt = 0; nt < 8; ++nt) {
    const int n = nt * 16 + n0;
    const float bias = be2[n];
    #pragma unroll
    for (int r = 0; r < 8; ++r) {
      const int m0 = mhi + r;
      const int m1 = 16 + mhi + r;
      s.u.M[wid][m0][n] = silu_f(acc2a[nt][r] + bias);
      s.u.M[wid][m1][n] = silu_f(acc2b[nt][r] + bias);
    }
  }
  __syncthreads();

  // ---- gate e_ij (one lane per edge) ----
  {
    float dot = 0.f;
    for (int c2 = 0; c2 < HID; ++c2) dot += s.u.M[wid][lane][c2] * Winf[c2];
    s.G[wid][lane] = 1.f / (1.f + __expf(-(dot + binf[0])));
  }
  __syncthreads();

  // ---- segment-sum scatter: m_i[src] += e_ij * m_ij ----
  for (int j = 0; j < 32; ++j) {
    long long e = e0 + j;
    if (e >= E_CNT) break;
    long long src = eidx[2 * e];
    float eg = s.G[wid][j];
    float* dstp = m_i + src * HID;
    #pragma unroll
    for (int q = 0; q < 4; ++q) {
      int n = lane * 4 + q;
      atomicAdd(dstp + n, eg * s.u.M[wid][j][n]);
    }
  }
}

// ---------------------------------------------------------------- node kernel
struct NodeSmem {
  alignas(16) bf16_t A[2][32][264];
  alignas(16) bf16_t H[2][32][136];
};

__global__ __launch_bounds__(64) void k_node(
    const bf16_t* __restrict__ xbf, const float* __restrict__ x,
    const float* __restrict__ m_i,
    const bf16_t* __restrict__ wh1p, const float* __restrict__ bh1,
    const bf16_t* __restrict__ wh2p, const float* __restrict__ bh2,
    float* __restrict__ out)
{
  __shared__ NodeSmem s;
  const int lane = threadIdx.x & 31;
  const int wid  = threadIdx.x >> 5;
  const long long i0 = ((long long)blockIdx.x * 2 + wid) * 32;

  for (int j = 0; j < 32; ++j) {
    long long node = i0 + j;
    if (node >= N_CNT) node = N_CNT - 1;
    const unsigned* px = (const unsigned*)(xbf + node * HID);
    unsigned* dst = (unsigned*)(&s.A[wid][j][0]);
    dst[lane * 2 + 0] = px[lane * 2 + 0];
    dst[lane * 2 + 1] = px[lane * 2 + 1];
    const float4 mv = *(const float4*)(m_i + node * HID + lane * 4);
    s.A[wid][j][HID + lane * 4 + 0] = (bf16_t)mv.x;
    s.A[wid][j][HID + lane * 4 + 1] = (bf16_t)mv.y;
    s.A[wid][j][HID + lane * 4 + 2] = (bf16_t)mv.z;
    s.A[wid][j][HID + lane * 4 + 3] = (bf16_t)mv.w;
  }
  __syncthreads();

  const int rlo  = lane & 15;
  const int koff = (lane >> 4) * 8;
  const int n0   = lane & 15;
  const int mhi  = (lane >> 4) * 8;
  const bf16_t* b1base = wh1p + lane * 16;
  const bf16_t* b2base = wh2p + lane * 16;

  v8f acc0[8], acc1[8];
  #pragma unroll
  for (int nt = 0; nt < 8; ++nt)
    #pragma unroll
    for (int r = 0; r < 8; ++r) { acc0[nt][r] = 0.f; acc1[nt][r] = 0.f; }

  {
    v16bf a0 = load_afrag(&s.A[wid][rlo][koff]);
    v16bf a1 = load_afrag(&s.A[wid][16 + rlo][koff]);
    v16bf bc = *(const v16bf*)(b1base);
    #pragma unroll
    for (int kc = 0; kc < 8; ++kc) {
      v16bf a0n = a0, a1n = a1;
      if (kc + 1 < 8) {
        a0n = load_afrag(&s.A[wid][rlo][(kc + 1) * 32 + koff]);
        a1n = load_afrag(&s.A[wid][16 + rlo][(kc + 1) * 32 + koff]);
      }
      #pragma unroll
      for (int nt = 0; nt < 8; ++nt) {
        const int idx = kc * 8 + nt;
        v16bf bn = bc;
        if (idx + 1 < 64) bn = *(const v16bf*)(b1base + (idx + 1) * 512);
        acc0[nt] = __builtin_amdgcn_wmma_f32_16x16x32_bf16(false, a0, false, bc,
                                                           (short)0, acc0[nt], false, false);
        acc1[nt] = __builtin_amdgcn_wmma_f32_16x16x32_bf16(false, a1, false, bc,
                                                           (short)0, acc1[nt], false, false);
        bc = bn;
      }
      a0 = a0n; a1 = a1n;
    }
  }
  __syncthreads();
  #pragma unroll
  for (int nt = 0; nt < 8; ++nt) {
    const int n = nt * 16 + n0;
    const float bias = bh1[n];
    #pragma unroll
    for (int r = 0; r < 8; ++r) {
      const int m0 = mhi + r;
      const int m1 = 16 + mhi + r;
      s.H[wid][m0][n] = (bf16_t)silu_f(acc0[nt][r] + bias);
      s.H[wid][m1][n] = (bf16_t)silu_f(acc1[nt][r] + bias);
    }
  }
  __syncthreads();

  v8f acc2a[8], acc2b[8];
  #pragma unroll
  for (int nt = 0; nt < 8; ++nt)
    #pragma unroll
    for (int r = 0; r < 8; ++r) { acc2a[nt][r] = 0.f; acc2b[nt][r] = 0.f; }

  {
    v16bf a0 = load_afrag(&s.H[wid][rlo][koff]);
    v16bf a1 = load_afrag(&s.H[wid][16 + rlo][koff]);
    v16bf bc = *(const v16bf*)(b2base);
    #pragma unroll
    for (int kc = 0; kc < 4; ++kc) {
      v16bf a0n = a0, a1n = a1;
      if (kc + 1 < 4) {
        a0n = load_afrag(&s.H[wid][rlo][(kc + 1) * 32 + koff]);
        a1n = load_afrag(&s.H[wid][16 + rlo][(kc + 1) * 32 + koff]);
      }
      #pragma unroll
      for (int nt = 0; nt < 8; ++nt) {
        const int idx = kc * 8 + nt;
        v16bf bn = bc;
        if (idx + 1 < 32) bn = *(const v16bf*)(b2base + (idx + 1) * 512);
        acc2a[nt] = __builtin_amdgcn_wmma_f32_16x16x32_bf16(false, a0, false, bc,
                                                            (short)0, acc2a[nt], false, false);
        acc2b[nt] = __builtin_amdgcn_wmma_f32_16x16x32_bf16(false, a1, false, bc,
                                                            (short)0, acc2b[nt], false, false);
        bc = bn;
      }
      a0 = a0n; a1 = a1n;
    }
  }
  #pragma unroll
  for (int nt = 0; nt < 8; ++nt) {
    const int n = nt * 16 + n0;
    const float bias = bh2[n];
    #pragma unroll
    for (int r = 0; r < 8; ++r) {
      const int m0 = mhi + r;
      const int m1 = 16 + mhi + r;
      long long node0 = i0 + m0;
      long long node1 = i0 + m1;
      if (node0 < N_CNT)
        out[node0 * HID + n] = x[node0 * HID + n] + acc2a[nt][r] + bias;
      if (node1 < N_CNT)
        out[node1 * HID + n] = x[node1 * HID + n] + acc2b[nt][r] + bias;
    }
  }
}

// ---------------------------------------------------------------- pos kernel
__global__ void k_pos(const float* __restrict__ pos, const float* __restrict__ mp,
                      const float* __restrict__ Whp1, const float* __restrict__ bhp1,
                      const float* __restrict__ Whp2, const float* __restrict__ bhp2,
                      float* __restrict__ outp)
{
  int i = blockIdx.x * blockDim.x + threadIdx.x;
  if (i >= N_CNT) return;
  float c[6];
  c[0] = pos[3 * i + 0]; c[1] = pos[3 * i + 1]; c[2] = pos[3 * i + 2];
  c[3] = mp[3 * i + 0];  c[4] = mp[3 * i + 1];  c[5] = mp[3 * i + 2];
  float h[3];
  #pragma unroll
  for (int p = 0; p < 3; ++p) {
    float a = bhp1[p];
    #pragma unroll
    for (int q = 0; q < 6; ++q) a += c[q] * Whp1[q * 3 + p];
    h[p] = silu_f(a);
  }
  #pragma unroll
  for (int p = 0; p < 3; ++p) {
    float a = bhp2[p];
    #pragma unroll
    for (int q = 0; q < 3; ++q) a += h[q] * Whp2[q * 3 + p];
    outp[3 * i + p] = c[p] + a;
  }
}

// ---------------------------------------------------------------- launch
extern "C" void kernel_launch(void* const* d_in, const int* in_sizes, int n_in,
                              void* d_out, int out_size, void* d_ws, size_t ws_size,
                              hipStream_t stream) {
  const float* x        = (const float*)d_in[0];
  const float* pos      = (const float*)d_in[1];
  const long long* eidx = (const long long*)d_in[2];
  const float* We1  = (const float*)d_in[3];
  const float* be1  = (const float*)d_in[4];
  const float* We2  = (const float*)d_in[5];
  const float* be2  = (const float*)d_in[6];
  const float* Winf = (const float*)d_in[7];
  const float* binf = (const float*)d_in[8];
  const float* Wh1  = (const float*)d_in[9];
  const float* bh1  = (const float*)d_in[10];
  const float* Wh2  = (const float*)d_in[11];
  const float* bh2  = (const float*)d_in[12];
  const float* Wp1  = (const float*)d_in[13];
  const float* bp1  = (const float*)d_in[14];
  const float* Wp2  = (const float*)d_in[15];
  const float* bp2  = (const float*)d_in[16];
  const float* Whp1 = (const float*)d_in[17];
  const float* bhp1 = (const float*)d_in[18];
  const float* Whp2 = (const float*)d_in[19];
  const float* bhp2 = (const float*)d_in[20];

  char* ws = (char*)d_ws;
  bf16_t* xbf     = (bf16_t*)(ws + 0);          // N*128 bf16 = 12,800,000 B
  bf16_t* we1p    = (bf16_t*)(ws + 12800000);   // 65,536 B
  bf16_t* we2p    = (bf16_t*)(ws + 12865536);   // 32,768 B
  bf16_t* wh1p    = (bf16_t*)(ws + 12898304);   // 65,536 B
  bf16_t* wh2p    = (bf16_t*)(ws + 12963840);   // 32,768 B
  float*  m_i     = (float*)(ws + 12996608);    // N*128 f32 = 25,600,000 B
  float*  m_i_pos = (float*)(ws + 38596608);    // N*3 f32, contiguous after m_i

  k_zero<<<2048, 256, 0, stream>>>(m_i, (long long)N_CNT * HID + (long long)N_CNT * 3);
  k_cvt_bf16<<<4096, 256, 0, stream>>>(x, xbf, (long long)N_CNT * HID);
  k_pack_b<<<(256 * 128 + 255) / 256, 256, 0, stream>>>(We1, we1p, 256);
  k_pack_b<<<(128 * 128 + 255) / 256, 256, 0, stream>>>(We2, we2p, 128);
  k_pack_b<<<(256 * 128 + 255) / 256, 256, 0, stream>>>(Wh1, wh1p, 256);
  k_pack_b<<<(128 * 128 + 255) / 256, 256, 0, stream>>>(Wh2, wh2p, 128);

  k_edge<<<E_CNT / 64, 64, 0, stream>>>(xbf, pos, eidx,
                                        we1p, We1, be1, we2p, be2,
                                        Winf, binf, Wp1, bp1, Wp2, bp2,
                                        m_i, m_i_pos);
  k_node<<<(N_CNT + 63) / 64, 64, 0, stream>>>(xbf, x, m_i, wh1p, bh1, wh2p, bh2,
                                               (float*)d_out);
  k_pos<<<(N_CNT + 255) / 256, 256, 0, stream>>>(pos, m_i_pos, Whp1, bhp1, Whp2, bhp2,
                                                 (float*)d_out + (long long)N_CNT * HID);
}
